// Edge_Conv_10299331576128
// MI455X (gfx1250) — matrix operand order repaired and checked by reading, not compiled, not run on hardware
//
#include <hip/hip_runtime.h>
#include <hip/hip_bf16.h>

typedef __attribute__((ext_vector_type(16))) _Float16 v16h;
typedef __attribute__((ext_vector_type(8)))  float    v8f;

#define N_CLOUDS 64
#define P_PTS    1024
#define CF       16
#define KNN      16
#define BN_EPS   1e-5f
#define WAVES    4

// ---------------- Kernel 1: 2D KNN (top-16, self excluded) + per-cloud valid count ----
__global__ __launch_bounds__(1024)
void knn_kernel(const float* __restrict__ features, const unsigned char* __restrict__ mask,
                int* __restrict__ idx_out, int* __restrict__ valid_out) {
  __shared__ float spts[P_PTS * 2];
  __shared__ int smask;
  const int n = blockIdx.x;
  const int q = threadIdx.x;
  const float* fbase = features + (size_t)n * P_PTS * (2 + CF);
  spts[q * 2 + 0] = fbase[q * (2 + CF) + 0];
  spts[q * 2 + 1] = fbase[q * (2 + CF) + 1];
  if (q == 0) smask = 0;
  __syncthreads();
  if (mask[n * P_PTS + q]) atomicAdd(&smask, 1);

  const float qx = spts[q * 2 + 0], qy = spts[q * 2 + 1];
  float bd[KNN]; int bi[KNN];
#pragma unroll
  for (int t = 0; t < KNN; ++t) { bd[t] = 3.0e38f; bi[t] = 0; }
  for (int c = 0; c < P_PTS; ++c) {
    if (c == q) continue;
    float dx = qx - spts[c * 2 + 0];
    float dy = qy - spts[c * 2 + 1];
    float d  = dx * dx + dy * dy;
    if (d >= bd[KNN - 1]) continue;
    int pos = 0;
#pragma unroll
    for (int t = 0; t < KNN; ++t) pos += (bd[t] <= d) ? 1 : 0;
    float cd = d; int ci = c;
#pragma unroll
    for (int t = 0; t < KNN; ++t) {
      bool ge = (t >= pos);
      float od = bd[t]; int oi = bi[t];
      bd[t] = ge ? cd : od;  bi[t] = ge ? ci : oi;
      cd    = ge ? od : cd;  ci    = ge ? oi : ci;
    }
  }
  int* out = idx_out + ((size_t)n * P_PTS + q) * KNN;
#pragma unroll
  for (int t = 0; t < KNN; ++t) out[t] = bi[t];
  __syncthreads();
  if (q == 0) valid_out[n] = P_PTS - smask;
}

// ---- A fragment (16x32 f16, ISA 7.12.2): lane l -> row m=l&15;
//      halves: K = kh..kh+7 and 16+kh..16+kh+7; kh = 8*(l>>4)
__device__ inline v16h load_A16(const _Float16* base, int koff, int lane) {
  const int m  = lane & 15;
  const int kh = (lane >> 4) << 3;
  const _Float16* p0 = base + m * 64 + koff + kh;
  v16h a;
#pragma unroll
  for (int i = 0; i < 8; ++i) { a[i] = p0[i]; a[8 + i] = p0[16 + i]; }
  return a;
}

// ---- B fragment (32x16 f16): lane l -> col n=l&15; halves: K = kb..kb+7 and
//      16+kb..16+kb+7, kb = 8*(l>>4).  B[k][n] = W[tile*16+n][koff+k].
__device__ inline v16h load_B16(const _Float16* W, int inc, int tile, int koff, int lane) {
  const int nn = lane & 15;
  const int kb = (lane >> 4) << 3;
  const _Float16* r = W + (tile * 16 + nn) * inc + koff + kb;
  v16h b;
#pragma unroll
  for (int i = 0; i < 8; ++i) { b[i] = r[i]; b[8 + i] = r[16 + i]; }
  return b;
}

// ---------------- Kernel 2: edge MLP via WMMA, masked mean, shortcut, relu ----------
__global__ __launch_bounds__(WAVES * 32)
void edgeconv_kernel(const float* __restrict__ features,
                     const unsigned char* __restrict__ mask,
                     const float* __restrict__ W0f, const float* __restrict__ W1f,
                     const float* __restrict__ W2f, const float* __restrict__ Wsf,
                     const float* __restrict__ g0, const float* __restrict__ b0,
                     const float* __restrict__ m0, const float* __restrict__ v0,
                     const float* __restrict__ g1, const float* __restrict__ b1,
                     const float* __restrict__ m1, const float* __restrict__ v1,
                     const float* __restrict__ g2, const float* __restrict__ b2,
                     const float* __restrict__ m2, const float* __restrict__ v2,
                     const float* __restrict__ gs, const float* __restrict__ bs,
                     const float* __restrict__ ms, const float* __restrict__ vs,
                     const int* __restrict__ idx_in, const int* __restrict__ valid_in,
                     float* __restrict__ out) {
  __shared__ _Float16 wlds[2048 + 4096 + 4096];   // W0 | W1 | W2 (row-major f16)
  __shared__ _Float16 bufA[WAVES][16 * 64];
  __shared__ _Float16 bufB[WAVES][16 * 64];
  __shared__ float xcs[WAVES][16];
  __shared__ float wms[WAVES][16];
  __shared__ float bnsc[4][64];                   // fused BN scale per layer (0,1,2,shortcut)
  __shared__ float bnsh[4][64];                   // fused BN shift per layer

  const int tid  = threadIdx.x;
  const int lane = tid & 31;
  const int wv   = tid >> 5;

  __builtin_prefetch(W1f, 0, 1);   // global_prefetch_b8: pull weights toward L2/L0

  // cooperative prologue: stage f16 weights + folded BN affines (once per block)
  for (int i = tid; i < 2048; i += WAVES * 32) wlds[i] = (_Float16)W0f[i];
  for (int i = tid; i < 4096; i += WAVES * 32) wlds[2048 + i] = (_Float16)W1f[i];
  for (int i = tid; i < 4096; i += WAVES * 32) wlds[2048 + 4096 + i] = (_Float16)W2f[i];
  {
    const float* gp[4] = { g0, g1, g2, gs };
    const float* bp[4] = { b0, b1, b2, bs };
    const float* mp[4] = { m0, m1, m2, ms };
    const float* vp[4] = { v0, v1, v2, vs };
    for (int i = tid; i < 4 * 64; i += WAVES * 32) {
      const int L = i >> 6, o = i & 63;
      float sc = gp[L][o] * rsqrtf(vp[L][o] + BN_EPS);
      bnsc[L][o] = sc;
      bnsh[L][o] = bp[L][o] - mp[L][o] * sc;
    }
  }
  __syncthreads();

  const int pid = blockIdx.x * WAVES + wv;      // point id 0..65535
  const int n   = pid >> 10;
  const int p   = pid & 1023;
  const int valid = valid_in[n];
  const float* fb   = features + (size_t)n * P_PTS * (2 + CF);
  const int*   nidx = idx_in + ((size_t)n * P_PTS + p) * KNN;

  // --- gather: h[j] = [xc, xc - x_neighbor(j)] (16 edges x 32 ch, f16) ---
  if (lane < 16) {
    const int j  = lane;
    const int nb = nidx[j];
    wms[wv][j] = (nb >= valid) ? 0.0f : 1.0f;        // mask_K: idx >= valid
    xcs[wv][j] = fb[p * (2 + CF) + 2 + j];
    _Float16* row = &bufA[wv][j * 64];
#pragma unroll
    for (int c = 0; c < CF; ++c) {
      float xc = fb[p  * (2 + CF) + 2 + c];
      float xn = fb[nb * (2 + CF) + 2 + c];
      row[c]      = (_Float16)xc;
      row[16 + c] = (_Float16)(xc - xn);
    }
  }
  __syncthreads();

  // --- Layer 0: (16x32) x W0^T -> 16x64, BN+ReLU -> bufB ---
  {
    v16h a = load_A16(&bufA[wv][0], 0, lane);
    const int mb = (lane >> 4) << 3;
#pragma unroll
    for (int t = 0; t < 4; ++t) {
      v8f c = {};
      v16h bmat = load_B16(&wlds[0], 32, t, 0, lane);
      c = __builtin_amdgcn_wmma_f32_16x16x32_f16(false, a, false, bmat, (short)0, c, false, false);
      const int o = t * 16 + (lane & 15);
      const float sc = bnsc[0][o], sh = bnsh[0][o];
#pragma unroll
      for (int r = 0; r < 8; ++r)
        bufB[wv][(mb + r) * 64 + o] = (_Float16)fmaxf(c[r] * sc + sh, 0.0f);
    }
  }
  __syncthreads();

  // --- Layer 1: (16x64) x W1^T -> 16x64, BN+ReLU -> bufA ---
  {
    v16h a0 = load_A16(&bufB[wv][0], 0, lane);
    v16h a1 = load_A16(&bufB[wv][0], 32, lane);
    const int mb = (lane >> 4) << 3;
#pragma unroll
    for (int t = 0; t < 4; ++t) {
      v8f c = {};
      v16h bb0 = load_B16(&wlds[2048], 64, t, 0, lane);
      v16h bb1 = load_B16(&wlds[2048], 64, t, 32, lane);
      c = __builtin_amdgcn_wmma_f32_16x16x32_f16(false, a0, false, bb0, (short)0, c, false, false);
      c = __builtin_amdgcn_wmma_f32_16x16x32_f16(false, a1, false, bb1, (short)0, c, false, false);
      const int o = t * 16 + (lane & 15);
      const float sc = bnsc[1][o], sh = bnsh[1][o];
#pragma unroll
      for (int r = 0; r < 8; ++r)
        bufA[wv][(mb + r) * 64 + o] = (_Float16)fmaxf(c[r] * sc + sh, 0.0f);
    }
  }
  __syncthreads();

  // --- Layer 2 + masked mean over K + shortcut + relu + transposed store ---
  {
    v16h a0 = load_A16(&bufA[wv][0], 0, lane);
    v16h a1 = load_A16(&bufA[wv][0], 32, lane);
    const int mb = (lane >> 4) << 3;
    float wloc[8];
#pragma unroll
    for (int r = 0; r < 8; ++r) wloc[r] = wms[wv][mb + r];
    float wsum = 0.f;
#pragma unroll
    for (int r = 0; r < 8; ++r) wsum += wloc[r];
    wsum += __shfl_xor(wsum, 16, 32);
    const float denom = fmaxf(wsum, 1e-8f);
    const bool pmask = mask[n * P_PTS + p] != 0;

#pragma unroll
    for (int t = 0; t < 4; ++t) {
      v8f c = {};
      v16h bb0 = load_B16(&wlds[2048 + 4096], 64, t, 0, lane);
      v16h bb1 = load_B16(&wlds[2048 + 4096], 64, t, 32, lane);
      c = __builtin_amdgcn_wmma_f32_16x16x32_f16(false, a0, false, bb0, (short)0, c, false, false);
      c = __builtin_amdgcn_wmma_f32_16x16x32_f16(false, a1, false, bb1, (short)0, c, false, false);
      const int o = t * 16 + (lane & 15);
      const float sc = bnsc[2][o], sh = bnsh[2][o];
      float part = 0.f;
#pragma unroll
      for (int r = 0; r < 8; ++r)
        part += wloc[r] * fmaxf(c[r] * sc + sh, 0.0f);
      part += __shfl_xor(part, 16, 32);   // combine M halves (lanes l and l+16)
      if (lane < 16) {
        float agg = part / denom;
        float s = 0.f;
#pragma unroll
        for (int cc = 0; cc < CF; ++cc) s += xcs[wv][cc] * Wsf[o * CF + cc];
        float scv = pmask ? 0.0f : (s * bnsc[3][o] + bnsh[3][o]);
        out[((size_t)n * 64 + o) * P_PTS + p] = fmaxf(agg + scv, 0.0f);
      }
    }
  }
}

extern "C" void kernel_launch(void* const* d_in, const int* in_sizes, int n_in,
                              void* d_out, int out_size, void* d_ws, size_t ws_size,
                              hipStream_t stream) {
  const float* features = (const float*)d_in[0];
  const unsigned char* mask = (const unsigned char*)d_in[1];
  const float* W0 = (const float*)d_in[2];
  const float* W1 = (const float*)d_in[3];
  const float* W2 = (const float*)d_in[4];
  const float* Ws = (const float*)d_in[5];
  const float *g0 = (const float*)d_in[6],  *b0 = (const float*)d_in[7],
              *m0 = (const float*)d_in[8],  *v0 = (const float*)d_in[9];
  const float *g1 = (const float*)d_in[10], *b1 = (const float*)d_in[11],
              *m1 = (const float*)d_in[12], *v1 = (const float*)d_in[13];
  const float *g2 = (const float*)d_in[14], *b2 = (const float*)d_in[15],
              *m2 = (const float*)d_in[16], *v2 = (const float*)d_in[17];
  const float *gs = (const float*)d_in[18], *bs = (const float*)d_in[19],
              *ms = (const float*)d_in[20], *vs = (const float*)d_in[21];

  int* idxws   = (int*)d_ws;                                   // N*P*K ints
  int* validws = idxws + (size_t)N_CLOUDS * P_PTS * KNN;       // N ints
  float* out   = (float*)d_out;

  knn_kernel<<<N_CLOUDS, P_PTS, 0, stream>>>(features, mask, idxws, validws);

  edgeconv_kernel<<<(N_CLOUDS * P_PTS) / WAVES, WAVES * 32, 0, stream>>>(
      features, mask, W0, W1, W2, Ws,
      g0, b0, m0, v0, g1, b1, m1, v1, g2, b2, m2, v2, gs, bs, ms, vs,
      idxws, validws, out);
}
